// Encoder_16741782520595
// MI455X (gfx1250) — compile-verified, matrix operand order checked by reading, not run on hardware
//
#include <hip/hip_runtime.h>
#include <hip/hip_bf16.h>

// ---------------------------------------------------------------------------
// GRU (B=64, T=256, U=1024, VOCAB=10000) for gfx1250 / MI455X.
//   rec = h @ W_rec via v_wmma_f32_16x16x32_bf16 with split-bf16 (hi+lo)
//   emulation of fp32:  A*B ~= AhiBhi + AloBhi + AhiBlo.
// W_rec is repacked once into the WMMA per-lane B-fragment layout (bf16 hi/lo),
// kept L2-resident (12.6 MB << 192 MB L2) across 256 sequential step kernels.
// ---------------------------------------------------------------------------

#define B_      64
#define T_      256
#define U_      1024
#define N3_     3072   // 3*U
#define KCHUNKS 32     // 1024 / 32
#define NTILES  192    // 3072 / 16

typedef __attribute__((ext_vector_type(16))) __bf16 v16bf;
typedef __attribute__((ext_vector_type(8)))  float  v8f;

// ---------------------------------------------------------------------------
// Repack W_rec (1024x3072 fp32, row-major [k][n]) into WMMA B-fragment order,
// split into bf16 hi/lo. Fragment (ntile, kc) lane l holds 16 bf16 contiguous:
//   slot s in 0..7 -> dword s holds K-pair (k, k+1) with
//   k = kc*32 + (s<4 ? 2s : 16+2(s-4)) + (l>=16 ? 8 : 0),  n = ntile*16 + (l&15)
// matching the ISA 16-bit B-matrix 32x16 VGPR layout.
// ---------------------------------------------------------------------------
__global__ void prep_wrec(const float* __restrict__ W_rec,
                          __bf16* __restrict__ WpH, __bf16* __restrict__ WpL) {
    int tid = blockIdx.x * blockDim.x + threadIdx.x;   // NTILES*KCHUNKS*32
    if (tid >= NTILES * KCHUNKS * 32) return;
    int lane  = tid & 31;
    int kc    = (tid >> 5) & 31;
    int ntile = tid >> 10;
    int n  = (ntile << 4) + (lane & 15);
    int kb = (lane >> 4) << 3;                  // 0 or 8
    size_t base = (size_t)tid * 16;
#pragma unroll
    for (int s = 0; s < 8; ++s) {
        int k = (kc << 5) + ((s < 4) ? (s << 1) : (16 + ((s - 4) << 1))) + kb;
#pragma unroll
        for (int p = 0; p < 2; ++p) {
            float w  = W_rec[(size_t)(k + p) * N3_ + n];
            __bf16 h = (__bf16)w;
            __bf16 l = (__bf16)(w - (float)h);
            WpH[base + s * 2 + p] = h;
            WpL[base + s * 2 + p] = l;
        }
    }
}

__device__ __forceinline__ float sigmoidf_(float v) {
    return 1.0f / (1.0f + __expf(-v));
}

// ---------------------------------------------------------------------------
// One GRU timestep. Grid: 64 blocks (4 M-tiles x 16 U-tiles), 128 thr (4 waves).
// Each wave computes one 16x16 h-tile: three gate accumulators (z, r, hh) over
// K=1024 (32 chunks x 3 split-WMMAs each), then the gate nonlinearity and
// h_new entirely in-wave. Embedding gather (W_in[x[b,t]] + b_in) fused into
// the epilogue.
// ---------------------------------------------------------------------------
__global__ __launch_bounds__(128) void gru_step(
    const int*    __restrict__ x,
    const float*  __restrict__ hin,      // (64,1024) fp32, h_{t-1}
    const float*  __restrict__ W_in,     // (VOCAB,3072)
    const __bf16* __restrict__ WpH,
    const __bf16* __restrict__ WpL,
    const float*  __restrict__ b_in,
    const float*  __restrict__ b_rec,
    float*        __restrict__ out,      // (64,256,1024)
    float*        __restrict__ hout,     // (64,1024) fp32, h_t
    int t)
{
    const int lane  = threadIdx.x & 31;
    const int wave  = threadIdx.x >> 5;
    const int mtile = blockIdx.x & 3;
    const int usub  = ((blockIdx.x >> 2) << 2) + wave;   // 0..63
    const int u0    = usub << 4;
    const int m0    = mtile << 4;
    const int lb    = lane & 15;
    const int kb    = (lane >> 4) << 3;                  // 0 or 8
    const int mrow  = m0 + lb;                           // A-matrix row

    v8f accz = {}, accr = {}, acch = {};
    const float* hrow = hin + (size_t)mrow * U_;

    for (int kc = 0; kc < KCHUNKS; ++kc) {
        // Build A fragments (h rows, fp32 -> bf16 hi/lo), ISA 16-bit A 16x32 layout.
        v16bf aH, aL;
#pragma unroll
        for (int s = 0; s < 8; ++s) {
            int k = (kc << 5) + ((s < 4) ? (s << 1) : (16 + ((s - 4) << 1))) + kb;
            float2 hv = *(const float2*)(hrow + k);
            __bf16 h0 = (__bf16)hv.x; __bf16 l0 = (__bf16)(hv.x - (float)h0);
            __bf16 h1 = (__bf16)hv.y; __bf16 l1 = (__bf16)(hv.y - (float)h1);
            aH[2 * s] = h0; aH[2 * s + 1] = h1;
            aL[2 * s] = l0; aL[2 * s + 1] = l1;
        }
#pragma unroll
        for (int g = 0; g < 3; ++g) {
            size_t ntile = (size_t)g * 64 + usub;
            size_t off   = (((ntile * KCHUNKS) + kc) * 32 + lane) * 16;
            v16bf bH = *(const v16bf*)(WpH + off);
            v16bf bL = *(const v16bf*)(WpL + off);
            v8f acc = (g == 0) ? accz : (g == 1) ? accr : acch;
            acc = __builtin_amdgcn_wmma_f32_16x16x32_bf16(false, aH, false, bH,
                                                          (short)0, acc, false, false);
            acc = __builtin_amdgcn_wmma_f32_16x16x32_bf16(false, aL, false, bH,
                                                          (short)0, acc, false, false);
            acc = __builtin_amdgcn_wmma_f32_16x16x32_bf16(false, aH, false, bL,
                                                          (short)0, acc, false, false);
            if (g == 0) accz = acc; else if (g == 1) accr = acc; else acch = acc;
        }
    }

    // Epilogue: C/D layout -> lane (0..15 | 16..31) holds col u0+(lane&15),
    // rows m0+i (lanes<16) / m0+8+i (lanes>=16) in acc[i].
    const int u     = u0 + lb;
    const int rbase = m0 + ((lane >> 4) << 3);
    const float brz = b_rec[u];
    const float brr = b_rec[U_ + u];
    const float brh = b_rec[2 * U_ + u];
    const float biz = b_in[u];
    const float bir = b_in[U_ + u];
    const float bih = b_in[2 * U_ + u];

#pragma unroll
    for (int i = 0; i < 8; ++i) {
        int b   = rbase + i;
        int tok = x[b * T_ + t];
        const float* wi = W_in + (size_t)tok * N3_;
        float xz = wi[u]           + biz;
        float xr = wi[U_ + u]      + bir;
        float xh = wi[2 * U_ + u]  + bih;
        float rz = accz[i] + brz;
        float rr = accr[i] + brr;
        float rh = acch[i] + brh;
        float z  = sigmoidf_(xz + rz);
        float r  = sigmoidf_(xr + rr);
        float hh = tanhf(xh + r * rh);
        float hp = hin[(size_t)b * U_ + u];
        float hn = z * hp + (1.0f - z) * hh;
        out[((size_t)b * T_ + t) * U_ + u] = hn;
        hout[(size_t)b * U_ + u]           = hn;
    }
}

__global__ void copy_state(const float* __restrict__ h, float* __restrict__ dst) {
    int i = blockIdx.x * blockDim.x + threadIdx.x;
    if (i < B_ * U_) dst[i] = h[i];
}

extern "C" void kernel_launch(void* const* d_in, const int* in_sizes, int n_in,
                              void* d_out, int out_size, void* d_ws, size_t ws_size,
                              hipStream_t stream) {
    const int*   x      = (const int*)  d_in[0];
    const float* hidden = (const float*)d_in[1];
    const float* W_in   = (const float*)d_in[2];
    const float* W_rec  = (const float*)d_in[3];
    const float* b_in   = (const float*)d_in[4];
    const float* b_rec  = (const float*)d_in[5];
    float* out = (float*)d_out;

    // Workspace: W_rec split-bf16 packed (2 x 6.29 MB) + h ping-pong (2 x 256 KB)
    __bf16* WpH = (__bf16*)d_ws;
    __bf16* WpL = WpH + (size_t)U_ * N3_;
    float*  hb0 = (float*)(WpL + (size_t)U_ * N3_);
    float*  hb1 = hb0 + B_ * U_;

    prep_wrec<<<(NTILES * KCHUNKS * 32 + 255) / 256, 256, 0, stream>>>(W_rec, WpH, WpL);

    const float* hin = hidden;
    for (int t = 0; t < T_; ++t) {
        float* ho = (t & 1) ? hb1 : hb0;
        gru_step<<<64, 128, 0, stream>>>(x, hin, W_in, WpH, WpL, b_in, b_rec,
                                         out, ho, t);
        hin = ho;
    }
    copy_state<<<(B_ * U_ + 255) / 256, 256, 0, stream>>>(hin, out + (size_t)B_ * T_ * U_);
}
